// Money_former_nGPT_11312943858159
// MI455X (gfx1250) — matrix operand order repaired and checked by reading, not compiled, not run on hardware
//
#include <hip/hip_runtime.h>
#include <hip/hip_bf16.h>
#include <math.h>

typedef __attribute__((ext_vector_type(16))) __bf16 v16bf;
typedef __attribute__((ext_vector_type(8)))  float  v8f;
typedef __attribute__((__vector_size__(16))) int    v4i;

#define WMMA_BF16(A, Bv, C) \
  __builtin_amdgcn_wmma_f32_16x16x32_bf16(false, (A), false, (Bv), (short)0, (C), false, false)

// ---- model constants ----
constexpr int kB = 2, kT = 256, kS = 6, kF = 64;
constexpr int kD = 512, kH = 8, kHD = 64, kDFF = 2048, kNL = 6, kP = 5;
constexpr int kSEQ = 257;                 // T + 1
constexpr int kLT  = 1542;                // SEQ_PER * S
constexpr int kM   = 3084;                // B * LT (GEMM rows)
constexpr int kKP  = 1568;                // key rows padded to 32 (49 tiles)
constexpr int kQP  = 1664;                // query rows padded to 128 (13 blocks)
constexpr float kSQRT_D = 22.62741699796952f;

// per-layer packed bf16 weight buffer offsets (elements). All stored TRANSPOSED (N x K)
constexpr size_t WQKV_OFF = 0;                     // 1536 x 512
constexpr size_t WO_OFF   = 786432;                // 512 x 512
constexpr size_t WIN_OFF  = 1048576;               // 4096 x 512
constexpr size_t WFF_OFF  = 3145728;               // 512 x 2048
constexpr size_t WL_TOTAL = 4194304;

union Frag { v16bf v; uint4 q[2]; };

// ---- CDNA5 async global->LDS copy (ASYNCcnt path), with sync fallback ----
#if defined(__has_builtin)
#if __has_builtin(__builtin_amdgcn_global_load_async_to_lds_b128)
#define HAS_ASYNC_LDS 1
#endif
#endif
#ifndef HAS_ASYNC_LDS
#define HAS_ASYNC_LDS 0
#endif

__device__ __forceinline__ void async_copy16(void* lds, const void* g) {
#if HAS_ASYNC_LDS
  __builtin_amdgcn_global_load_async_to_lds_b128(
      (__attribute__((address_space(1))) v4i*)(void*)g,
      (__attribute__((address_space(3))) v4i*)lds, 0, 0);
#else
  *(uint4*)lds = *(const uint4*)g;
#endif
}
__device__ __forceinline__ void async_wait_all() {
#if HAS_ASYNC_LDS
  asm volatile("s_wait_asynccnt 0x0" ::: "memory");
#endif
}

__device__ __forceinline__ float blockSum256(float v, float* sh) {
  #pragma unroll
  for (int off = 16; off >= 1; off >>= 1) v += __shfl_xor(v, off, 32);
  __syncthreads();
  if ((threadIdx.x & 31) == 0) sh[threadIdx.x >> 5] = v;
  __syncthreads();
  float t = 0.f;
  #pragma unroll
  for (int i = 0; i < 8; i++) t += sh[i];
  return t;
}

// ===================== WMMA GEMM: C[M,N] = A[M,K] @ Bt[N,K]^T =====================
// A bf16 row-major, Bt bf16 N-major (pre-transposed weights), C f32.
// Block tile 128x128, K-step 32. 8 waves in 4(M) x 2(N); each wave owns 32x64:
// 2 A-frags x 4 B-frags -> 8 WMMAs per k-step (each B frag feeds 2 WMMAs).
__global__ __launch_bounds__(256) void gemm_bf16(const __bf16* __restrict__ A,
                                                 const __bf16* __restrict__ Bt,
                                                 float* __restrict__ C,
                                                 int M, int N, int K) {
  __shared__ __bf16 As[128][40];   // 80B row stride (16B aligned)
  __shared__ __bf16 Bs[128][40];   // Bs[n][k]
  const int tid = threadIdx.x;
  const int wv = tid >> 5, lane = tid & 31;
  const int col = lane & 15, khalf = lane >> 4;
  const int mw = wv & 3, nw = wv >> 2;
  const int m0 = blockIdx.y * 128, n0 = blockIdx.x * 128;

  v8f acc[2][4];
  #pragma unroll
  for (int a = 0; a < 2; a++)
    #pragma unroll
    for (int nc = 0; nc < 4; nc++) acc[a][nc] = (v8f){0,0,0,0,0,0,0,0};

  for (int k0 = 0; k0 < K; k0 += 32) {
    // stage A tile (128x32): 512 b128 chunks over 256 threads (row guard -> zero)
    #pragma unroll
    for (int c = tid; c < 512; c += 256) {
      const int r = c >> 2, k8 = (c & 3) << 3;
      const int gr = m0 + r;
      if (gr < M) async_copy16(&As[r][k8], A + (size_t)gr * K + k0 + k8);
      else        *(uint4*)(&As[r][k8]) = (uint4){0u, 0u, 0u, 0u};
    }
    // stage B tile (128 n x 32 k) straight from pre-transposed weights
    #pragma unroll
    for (int c = tid; c < 512; c += 256) {
      const int r = c >> 2, k8 = (c & 3) << 3;
      async_copy16(&Bs[r][k8], Bt + (size_t)(n0 + r) * K + k0 + k8);
    }
    if (k0 + 32 < K)   // pull next B k-tile toward L2/WGP$
      __builtin_prefetch(Bt + (size_t)(n0 + (tid >> 1)) * K + k0 + 32, 0, 1);
    async_wait_all();
    __syncthreads();

    // fragments (CDNA5 16-bit A 16x32 / B 32x16 lane layouts), then WMMA chain
    Frag a0, a1, bf[4];
    {
      const __bf16* ap0 = &As[mw * 32 + col][0];
      const __bf16* ap1 = &As[mw * 32 + 16 + col][0];
      a0.q[0] = *(const uint4*)(ap0 + khalf * 8);
      a0.q[1] = *(const uint4*)(ap0 + 16 + khalf * 8);
      a1.q[0] = *(const uint4*)(ap1 + khalf * 8);
      a1.q[1] = *(const uint4*)(ap1 + 16 + khalf * 8);
      #pragma unroll
      for (int nc = 0; nc < 4; nc++) {
        const __bf16* bp = &Bs[nw * 64 + nc * 16 + col][0];
        bf[nc].q[0] = *(const uint4*)(bp + khalf * 8);
        bf[nc].q[1] = *(const uint4*)(bp + 16 + khalf * 8);
      }
    }
    #pragma unroll
    for (int nc = 0; nc < 4; nc++) {
      acc[0][nc] = WMMA_BF16(a0.v, bf[nc].v, acc[0][nc]);
      acc[1][nc] = WMMA_BF16(a1.v, bf[nc].v, acc[1][nc]);
    }
    __syncthreads();
  }
  #pragma unroll
  for (int a = 0; a < 2; a++) {
    #pragma unroll
    for (int nc = 0; nc < 4; nc++) {
      const int gc = n0 + nw * 64 + nc * 16 + col;
      #pragma unroll
      for (int r = 0; r < 8; r++) {
        const int gr = m0 + mw * 32 + a * 16 + r + 8 * khalf;
        if (gr < M) C[(size_t)gr * N + gc] = acc[a][nc][r];
      }
    }
  }
}

// ============ per-layer weight pack f32 -> bf16, TRANSPOSED to N x K ============
__global__ __launch_bounds__(256) void pack_weights(const float* __restrict__ Wq,
    const float* __restrict__ Wk, const float* __restrict__ Wv,
    const float* __restrict__ Wo, const float* __restrict__ Win,
    const float* __restrict__ Wff, __bf16* __restrict__ Wl) {
  size_t idx = (size_t)blockIdx.x * 256 + threadIdx.x;
  if (idx >= WL_TOTAL) return;
  float v;
  if (idx < WO_OFF) {                       // QKV^T: (1536 x 512)
    const size_t n = idx / 512, k = idx % 512;
    const float* src = (n < 512) ? Wq : (n < 1024 ? Wk : Wv);
    v = src[k * 512 + (n & 511)];
  } else if (idx < WIN_OFF) {               // Wo^T: (512 x 512)
    const size_t j = idx - WO_OFF;
    v = Wo[(j % 512) * 512 + (j / 512)];
  } else if (idx < WFF_OFF) {               // Win^T: (4096 x 512)
    const size_t j = idx - WIN_OFF;
    v = Win[(j % 512) * 4096 + (j / 512)];
  } else {                                  // Wff^T: (512 x 2048)
    const size_t j = idx - WFF_OFF;
    v = Wff[(j % 2048) * 512 + (j / 2048)];
  }
  Wl[idx] = (__bf16)v;
}

// ===================== embedding + cosine norm =====================
__global__ __launch_bounds__(256) void embed_kernel(const float* __restrict__ x,
    const int* __restrict__ sep_idx, const int* __restrict__ tickers,
    const float* __restrict__ sep_emb, const float* __restrict__ tick_emb,
    const float* __restrict__ shared_W, const float* __restrict__ unique_W,
    float* __restrict__ hbuf, __bf16* __restrict__ Hb) {
  __shared__ float xs[64];
  __shared__ float sh[8];
  const int row = blockIdx.x;
  const int b = row / kLT, lin = row % kLT;
  const int t = lin / kS, s = lin % kS;
  if (t >= 1 && threadIdx.x < 64)
    xs[threadIdx.x] = x[(((size_t)b * kT + (t - 1)) * kS + s) * kF + threadIdx.x];
  __syncthreads();
  const int tk = tickers[b * kS + s];
  float vals[2];
  #pragma unroll
  for (int j = 0; j < 2; j++) {
    const int d = threadIdx.x + j * 256;
    float v;
    if (t == 0) {
      v = sep_emb[(size_t)sep_idx[b] * kD + d];
    } else if (d < 384) {
      v = 0.f;
      for (int f = 0; f < kF; f++) v += xs[f] * shared_W[(size_t)f * 384 + d];
    } else {
      v = 0.f;
      const int du = d - 384;
      for (int f = 0; f < kF; f++) v += xs[f] * unique_W[((size_t)s * kF + f) * 128 + du];
    }
    v += tick_emb[(size_t)tk * kD + d];
    vals[j] = v;
  }
  const float nn = blockSum256(vals[0] * vals[0] + vals[1] * vals[1], sh);
  const float inv = 1.0f / fmaxf(sqrtf(nn), 1e-6f);
  #pragma unroll
  for (int j = 0; j < 2; j++) {
    const int d = threadIdx.x + j * 256;
    const float v = vals[j] * inv;
    hbuf[(size_t)row * kD + d] = v;
    Hb[(size_t)row * kD + d] = (__bf16)v;
  }
}

// ======= QKV post: RoPE (YaRN freqs) + per-head cosine norm + reorder =======
// One wave per head; lane i owns the (2i, 2i+1) rotation pair.
// V is written TRANSPOSED per head: Vt[bh][hd][key(t-major)].
__global__ __launch_bounds__(256) void qkv_post(const float* __restrict__ QKV,
    const float* __restrict__ s_qk, __bf16* __restrict__ Qh,
    __bf16* __restrict__ Kh, __bf16* __restrict__ Vt) {
  const int row = blockIdx.x;
  const int b = row / kLT, lin = row % kLT;
  const int t = lin / kS, s = lin % kS;
  const int h = threadIdx.x >> 5, lane = threadIdx.x & 31;
  const float* base = QKV + (size_t)row * 1536 + h * kHD + lane * 2;
  float q0 = base[0], q1 = base[1];
  float k0 = base[512], k1 = base[513];
  const float v0 = base[1024], v1 = base[1025];
  if (t >= 1) {
    const double PI2 = 6.283185307179586476925286766559;
    const double i = (double)lane;
    double fr = pow(10000.0, -(2.0 * i) / 64.0);
    const double low  = fmax(floor(64.0 * log(256.0 / (32.0 * PI2)) / (2.0 * log(10000.0))), 0.0);
    double high = fmin(ceil(64.0 * log(256.0 / (1.0 * PI2)) / (2.0 * log(10000.0))), 63.0);
    const double hi = (low == high) ? high + 0.001 : high;
    const double ramp = fmin(fmax((i - low) / (hi - low), 0.0), 1.0);
    fr = fr / 40.0 * ramp + fr * (1.0 - ramp);
    const double ang = (double)(t - 1) * fr;
    const float c = (float)cos(ang), sn = (float)sin(ang);
    float nq0 = q0 * c - q1 * sn, nq1 = q0 * sn + q1 * c;
    float nk0 = k0 * c - k1 * sn, nk1 = k0 * sn + k1 * c;
    q0 = nq0; q1 = nq1; k0 = nk0; k1 = nk1;
  }
  float sq = q0 * q0 + q1 * q1;
  float sk = k0 * k0 + k1 * k1;
  #pragma unroll
  for (int off = 16; off >= 1; off >>= 1) {
    sq += __shfl_xor(sq, off, 32);
    sk += __shfl_xor(sk, off, 32);
  }
  const float invq = 1.0f / fmaxf(sqrtf(sq), 1e-6f);
  const float invk = 1.0f / fmaxf(sqrtf(sk), 1e-6f);
  const float HDIS = 0.125f;   // HD^-0.5
  const float HDS  = 8.0f;     // sqrt(HD), folded into q-hat
  const float e0 = s_qk[h * kHD + lane * 2] * HDIS;
  const float e1 = s_qk[h * kHD + lane * 2 + 1] * HDIS;
  const int bh = b * kH + h;
  const int srow = s * kSEQ + t;     // s-major position for Q/K
  __bf16* qp = Qh + ((size_t)bh * kQP + srow) * kHD + lane * 2;
  qp[0] = (__bf16)(q0 * invq * e0 * HDS);
  qp[1] = (__bf16)(q1 * invq * e1 * HDS);
  __bf16* kp = Kh + ((size_t)bh * kKP + srow) * kHD + lane * 2;
  kp[0] = (__bf16)(k0 * invk * e0);
  kp[1] = (__bf16)(k1 * invk * e1);
  Vt[((size_t)bh * kHD + lane * 2) * kKP + lin]     = (__bf16)v0;
  Vt[((size_t)bh * kHD + lane * 2 + 1) * kKP + lin] = (__bf16)v1;
}

// ===================== flash attention (WMMA, online softmax) =====================
__global__ __launch_bounds__(256) void flash_attn(const __bf16* __restrict__ Qh,
                                                  const __bf16* __restrict__ Kh,
                                                  const __bf16* __restrict__ Vt,
                                                  __bf16* __restrict__ Ob) {
  __shared__ __bf16 Ks[32][72];        // key tile row-major (keys x 64)
  __shared__ __bf16 Vs[64][40];        // value tile (hd x keys), already transposed in global
  __shared__ __bf16 Ps[8][16][40];     // per-wave P tile (16 q x 32 k)
  const int tid = threadIdx.x;
  const int wv = tid >> 5, lane = tid & 31;
  const int col = lane & 15, khalf = lane >> 4;
  const int bh = blockIdx.y;
  const int b = bh / kH, h = bh % kH;
  const int q0 = blockIdx.x * 128 + wv * 16;

  Frag aq0, aq1;   // Q fragments for kdim 0..31 and 32..63 (loop invariant)
  {
    const __bf16* qp = Qh + ((size_t)bh * kQP + q0 + col) * kHD;
    aq0.q[0] = *(const uint4*)(qp + khalf * 8);
    aq0.q[1] = *(const uint4*)(qp + 16 + khalf * 8);
    aq1.q[0] = *(const uint4*)(qp + 32 + khalf * 8);
    aq1.q[1] = *(const uint4*)(qp + 48 + khalf * 8);
  }
  v8f acc[4];
  #pragma unroll
  for (int i = 0; i < 4; i++) acc[i] = (v8f){0,0,0,0,0,0,0,0};
  float mrun[8], lrun[8];
  int tq[8];
  #pragma unroll
  for (int r = 0; r < 8; r++) {
    mrun[r] = -1e30f; lrun[r] = 0.f;
    tq[r] = (q0 + r + 8 * khalf) % kSEQ;
  }

  for (int kt = 0; kt < kKP / 32; kt++) {
    const int k0 = kt * 32;
    {
      const int kk = tid >> 3, n8 = (tid & 7) << 3;     // K tile: 32 x 64
      async_copy16(&Ks[kk][n8], Kh + ((size_t)bh * kKP + k0 + kk) * kHD + n8);
      const int vr = tid >> 2, vk8 = (tid & 3) << 3;    // V tile: 64 x 32 (hd-major)
      async_copy16(&Vs[vr][vk8], Vt + ((size_t)bh * kHD + vr) * kKP + k0 + vk8);
    }
    async_wait_all();
    __syncthreads();

    v8f sc[2];
    #pragma unroll
    for (int kc = 0; kc < 2; kc++) {
      Frag b0, b1;
      const __bf16* kp = &Ks[kc * 16 + col][0];
      b0.q[0] = *(const uint4*)(kp + khalf * 8);
      b0.q[1] = *(const uint4*)(kp + 16 + khalf * 8);
      b1.q[0] = *(const uint4*)(kp + 32 + khalf * 8);
      b1.q[1] = *(const uint4*)(kp + 48 + khalf * 8);
      v8f s = (v8f){0,0,0,0,0,0,0,0};
      s = WMMA_BF16(aq0.v, b0.v, s);
      s = WMMA_BF16(aq1.v, b1.v, s);
      sc[kc] = s;
    }
    // block-causal mask (assignment, so padded-row garbage cannot leak)
    #pragma unroll
    for (int kc = 0; kc < 2; kc++) {
      const int key = k0 + kc * 16 + col;
      const int tk = key % kSEQ;
      #pragma unroll
      for (int r = 0; r < 8; r++)
        if (key >= kLT || tk > tq[r]) sc[kc][r] = -1e30f;
    }
    // online softmax: 16-lane row reductions
    #pragma unroll
    for (int r = 0; r < 8; r++) {
      float mx = fmaxf(sc[0][r], sc[1][r]);
      #pragma unroll
      for (int off = 8; off >= 1; off >>= 1) mx = fmaxf(mx, __shfl_xor(mx, off, 16));
      const float mnew = fmaxf(mrun[r], mx);
      const float alpha = __expf(mrun[r] - mnew);
      const float p0 = __expf(sc[0][r] - mnew);
      const float p1 = __expf(sc[1][r] - mnew);
      float sum = p0 + p1;
      #pragma unroll
      for (int off = 8; off >= 1; off >>= 1) sum += __shfl_xor(sum, off, 16);
      lrun[r] = lrun[r] * alpha + sum;
      mrun[r] = mnew;
      #pragma unroll
      for (int nc = 0; nc < 4; nc++) acc[nc][r] = acc[nc][r] * alpha;
      Ps[wv][r + 8 * khalf][col] = (__bf16)p0;
      Ps[wv][r + 8 * khalf][16 + col] = (__bf16)p1;
    }
    asm volatile("s_wait_dscnt 0x0" ::: "memory");   // P stores -> P fragment loads (same wave)
    Frag pf;
    const __bf16* pp = &Ps[wv][col][0];
    pf.q[0] = *(const uint4*)(pp + khalf * 8);
    pf.q[1] = *(const uint4*)(pp + 16 + khalf * 8);
    #pragma unroll
    for (int nc = 0; nc < 4; nc++) {
      Frag bv;
      const __bf16* vp = &Vs[nc * 16 + col][0];
      bv.q[0] = *(const uint4*)(vp + khalf * 8);
      bv.q[1] = *(const uint4*)(vp + 16 + khalf * 8);
      acc[nc] = WMMA_BF16(pf.v, bv.v, acc[nc]);
    }
    __syncthreads();
  }
  #pragma unroll
  for (int r = 0; r < 8; r++) {
    const int qr = q0 + r + 8 * khalf;
    if (qr < kLT) {
      const float inv = 1.0f / lrun[r];
      #pragma unroll
      for (int nc = 0; nc < 4; nc++)
        Ob[((size_t)b * kLT + qr) * kD + h * kHD + nc * 16 + col] = (__bf16)(acc[nc][r] * inv);
    }
  }
}

// ============= residual eigen-lerp + double cosine norm =============
__global__ __launch_bounds__(256) void resid_norm(float* __restrict__ hbuf,
    __bf16* __restrict__ Hb, const float* __restrict__ Xf,
    const float* __restrict__ eigen) {
  __shared__ float sh[8];
  const int row = blockIdx.x;
  const int d0 = threadIdx.x, d1 = threadIdx.x + 256;
  const size_t base = (size_t)row * kD;
  const float a0 = Xf[base + d0], a1 = Xf[base + d1];
  const float na = blockSum256(a0 * a0 + a1 * a1, sh);
  const float inva = 1.0f / fmaxf(sqrtf(na), 1e-6f);
  const float h0 = hbuf[base + d0], h1 = hbuf[base + d1];
  const float e0 = eigen[d0] * kSQRT_D, e1 = eigen[d1] * kSQRT_D;
  float n0 = h0 + e0 * (a0 * inva - h0);
  float n1 = h1 + e1 * (a1 * inva - h1);
  const float nn = blockSum256(n0 * n0 + n1 * n1, sh);
  const float invn = 1.0f / fmaxf(sqrtf(nn), 1e-6f);
  n0 *= invn; n1 *= invn;
  hbuf[base + d0] = n0; hbuf[base + d1] = n1;
  Hb[base + d0] = (__bf16)n0; Hb[base + d1] = (__bf16)n1;
}

// ============= gated SiLU for FFN =============
__global__ __launch_bounds__(256) void ffn_act(const float* __restrict__ G,
    const float* __restrict__ s_u, const float* __restrict__ s_v,
    __bf16* __restrict__ Ab) {
  const int idx = blockIdx.x * 256 + threadIdx.x;
  if (idx >= kM * kDFF) return;
  const int row = idx / kDFF, j = idx % kDFF;
  const float u = G[(size_t)row * (2 * kDFF) + j];
  const float v = G[(size_t)row * (2 * kDFF) + kDFF + j];
  const float sv = v * s_v[j] * kSQRT_D;
  const float sig = 1.0f / (1.0f + __expf(-sv));
  Ab[idx] = (__bf16)(u * s_u[j] * (sv * sig));
}

// ============= output head (P=5) =============
__global__ __launch_bounds__(256) void out_proj(const float* __restrict__ hbuf,
    const float* __restrict__ Wout, const float* __restrict__ s_z,
    float* __restrict__ out) {
  __shared__ float sh[8];
  __shared__ float hrow[kD];
  const int row = blockIdx.x;
  hrow[threadIdx.x] = hbuf[(size_t)row * kD + threadIdx.x];
  hrow[threadIdx.x + 256] = hbuf[(size_t)row * kD + threadIdx.x + 256];
  __syncthreads();
  for (int p = 0; p < kP; p++) {
    float acc = 0.f;
    for (int d = threadIdx.x; d < kD; d += 256) acc += hrow[d] * Wout[(size_t)d * kP + p];
    const float tot = blockSum256(acc, sh);
    if (threadIdx.x == 0) out[(size_t)row * kP + p] = tot * s_z[p] * kSQRT_D;
  }
}

// ===================== host orchestration =====================
extern "C" void kernel_launch(void* const* d_in, const int* in_sizes, int n_in,
                              void* d_out, int out_size, void* d_ws, size_t ws_size,
                              hipStream_t stream) {
  (void)in_sizes; (void)n_in; (void)out_size; (void)ws_size;
  const float* x        = (const float*)d_in[0];
  const int*   sep      = (const int*)d_in[1];
  const int*   tickers  = (const int*)d_in[2];
  const float* sep_emb  = (const float*)d_in[3];
  const float* tick_emb = (const float*)d_in[4];
  const float* shared_W = (const float*)d_in[5];
  const float* unique_W = (const float*)d_in[6];
  const float* Wq       = (const float*)d_in[7];
  const float* Wk       = (const float*)d_in[8];
  const float* Wv       = (const float*)d_in[9];
  const float* Wo       = (const float*)d_in[10];
  const float* s_qk     = (const float*)d_in[11];
  const float* Win      = (const float*)d_in[12];
  const float* Wff      = (const float*)d_in[13];
  const float* s_u      = (const float*)d_in[14];
  const float* s_v      = (const float*)d_in[15];
  const float* eigen_a  = (const float*)d_in[16];
  const float* eigen_m  = (const float*)d_in[17];
  const float* Wout     = (const float*)d_in[18];
  const float* s_z      = (const float*)d_in[19];
  float* out = (float*)d_out;

  char* ws = (char*)d_ws;
  auto alloc = [&](size_t bytes) -> char* {
    char* p = ws;
    ws += (bytes + 255) & ~(size_t)255;
    return p;
  };
  __bf16* Wl   = (__bf16*)alloc(WL_TOTAL * 2);                       // packed layer weights (N x K)
  float*  hbuf = (float*) alloc((size_t)kM * kD * 4);                // fp32 residual state
  __bf16* Hb   = (__bf16*)alloc((size_t)kM * kD * 2);                // bf16 mirror (GEMM A)
  float*  BIG  = (float*) alloc((size_t)kM * 2 * kDFF * 4);          // QKV (Mx1536) / FFN pre-act (Mx4096)
  __bf16* Qh   = (__bf16*)alloc((size_t)kB * kH * kQP * kHD * 2);
  __bf16* Kh   = (__bf16*)alloc((size_t)kB * kH * kKP * kHD * 2);
  __bf16* Vt   = (__bf16*)alloc((size_t)kB * kH * kHD * kKP * 2);    // per-head V, hd-major
  __bf16* Ob   = (__bf16*)alloc((size_t)kM * kD * 2);
  float*  Af   = (float*) alloc((size_t)kM * kD * 4);
  __bf16* Ab   = (__bf16*)alloc((size_t)kM * kDFF * 2);

  const dim3 blk(256);
  embed_kernel<<<kM, blk, 0, stream>>>(x, sep, tickers, sep_emb, tick_emb,
                                       shared_W, unique_W, hbuf, Hb);
  for (int l = 0; l < kNL; l++) {
    pack_weights<<<(unsigned)((WL_TOTAL + 255) / 256), blk, 0, stream>>>(
        Wq + (size_t)l * kD * kD, Wk + (size_t)l * kD * kD, Wv + (size_t)l * kD * kD,
        Wo + (size_t)l * kD * kD, Win + (size_t)l * kD * 2 * kDFF,
        Wff + (size_t)l * kDFF * kD, Wl);
    // fused QKV projection: (M x 512) @ (512 x 1536)
    gemm_bf16<<<dim3(1536 / 128, (kM + 127) / 128), blk, 0, stream>>>(
        Hb, Wl + WQKV_OFF, BIG, kM, 1536, kD);
    qkv_post<<<kM, blk, 0, stream>>>(BIG, s_qk + (size_t)l * kH * kHD, Qh, Kh, Vt);
    flash_attn<<<dim3((kLT + 127) / 128, kB * kH), blk, 0, stream>>>(Qh, Kh, Vt, Ob);
    // attention output projection: (M x 512) @ (512 x 512)
    gemm_bf16<<<dim3(kD / 128, (kM + 127) / 128), blk, 0, stream>>>(
        Ob, Wl + WO_OFF, Af, kM, kD, kD);
    resid_norm<<<kM, blk, 0, stream>>>(hbuf, Hb, Af, eigen_a + (size_t)l * kD);
    // FFN in: (M x 512) @ (512 x 4096)
    gemm_bf16<<<dim3(2 * kDFF / 128, (kM + 127) / 128), blk, 0, stream>>>(
        Hb, Wl + WIN_OFF, BIG, kM, 2 * kDFF, kD);
    ffn_act<<<(kM * kDFF + 255) / 256, blk, 0, stream>>>(
        BIG, s_u + (size_t)l * kDFF, s_v + (size_t)l * kDFF, Ab);
    // FFN out: (M x 2048) @ (2048 x 512)
    gemm_bf16<<<dim3(kD / 128, (kM + 127) / 128), blk, 0, stream>>>(
        Ab, Wl + WFF_OFF, Af, kM, kD, kDFF);
    resid_norm<<<kM, blk, 0, stream>>>(hbuf, Hb, Af, eigen_m + (size_t)l * kD);
  }
  out_proj<<<kM, blk, 0, stream>>>(hbuf, Wout, s_z, out);
}
